// AttributeDecoder_2637109920182
// MI455X (gfx1250) — matrix operand order repaired and checked by reading, not compiled, not run on hardware
//
#include <hip/hip_runtime.h>
#include <hip/hip_bf16.h>

typedef _Float16 half_t;
typedef __attribute__((ext_vector_type(16))) _Float16 v16h;
typedef __attribute__((ext_vector_type(8)))  float    v8f;
typedef __attribute__((ext_vector_type(4)))  float    v4f;

#define N_NODES 8192
#define D_IN    64
#define D_HID   128
#define D_OUTC  256

// ---------------------------------------------------------------------------
// Kernel 1: T1T[c][r] = sum_k X[r][k] * W1[k][c], stored fp16 TRANSPOSED.
// ---------------------------------------------------------------------------
__global__ void k_xw1(const float* __restrict__ X, const float* __restrict__ W1,
                      half_t* __restrict__ T1T) {
  int idx = blockIdx.x * blockDim.x + threadIdx.x;   // N_NODES * D_HID threads
  int c = idx & (D_HID - 1);
  int r = idx >> 7;                                  // / D_HID
  float acc = 0.f;
#pragma unroll 8
  for (int k = 0; k < D_IN; ++k)
    acc += X[r * D_IN + k] * W1[k * D_HID + c];
  T1T[(size_t)c * N_NODES + r] = (half_t)acc;
}

// ---------------------------------------------------------------------------
// Kernel 3: T2T[c][r] = sum_k H1[r][k] * W2[k][c], fp16 transposed.
// ---------------------------------------------------------------------------
__global__ void k_hw2(const half_t* __restrict__ H1, const float* __restrict__ W2,
                      half_t* __restrict__ T2T) {
  int idx = blockIdx.x * blockDim.x + threadIdx.x;   // N_NODES * D_OUTC threads
  int c = idx & (D_OUTC - 1);
  int r = idx >> 8;                                  // / D_OUTC
  float acc = 0.f;
#pragma unroll 8
  for (int k = 0; k < D_HID; ++k)
    acc += (float)H1[r * D_HID + k] * W2[k * D_OUTC + c];
  T2T[(size_t)c * N_NODES + r] = (half_t)acc;
}

// ---------------------------------------------------------------------------
// Propagation: OUT = relu(adj @ B), B transposed fp16 [DOUT][N].
// Block = 256 threads = 8 waves = ONE 16-row strip, split-K 8 ways.
// Deterministic LDS tree reduction combines the 8 partials; wave 0 stores.
//
// Inner loop: batch B-loads in groups of 4 (bounded register pressure, no
// spills) with a sched fence so the 4 WMMAs drain staggered loadcnt waits.
// ---------------------------------------------------------------------------
template <int DOUT, bool STORE_HALF>
__global__ void __launch_bounds__(256, 1)
k_prop(const float* __restrict__ adj,
       const half_t* __restrict__ BT,
       void* __restrict__ outp) {
  constexpr int NT     = DOUT / 16;     // 8 or 16 column tiles
  constexpr int KCHUNK = N_NODES / 8;   // 1024 per wave

  __shared__ float red[4 * 32 * NT * 8];  // 32 KB (DOUT=128) / 64 KB (DOUT=256)

  const int lane = threadIdx.x & 31;
  // wave index is wave-uniform: force it scalar so loop control stays SALU
  const int wave = __builtin_amdgcn_readfirstlane(threadIdx.x >> 5);
  const int row0 = blockIdx.x * 16;
  const int m  = lane & 15;
  const int hi = lane >> 4;

  v8f acc[NT] = {};

  const float* arow = adj + (size_t)(row0 + m) * N_NODES + hi * 16;
  const int kbeg = wave * KCHUNK;
  const int kend = kbeg + KCHUNK;

  for (int k0 = kbeg; k0 < kend; k0 += 32) {
    // ---- A fragment: 4x16B nontemporal loads of streamed adj, cvt to f16
    v4f a0 = __builtin_nontemporal_load((const v4f*)(arow + k0 + 0));
    v4f a1 = __builtin_nontemporal_load((const v4f*)(arow + k0 + 4));
    v4f a2 = __builtin_nontemporal_load((const v4f*)(arow + k0 + 8));
    v4f a3 = __builtin_nontemporal_load((const v4f*)(arow + k0 + 12));
    v16h a;
#pragma unroll
    for (int i = 0; i < 4; ++i) {
      a[i]      = (half_t)a0[i];
      a[4 + i]  = (half_t)a1[i];
      a[8 + i]  = (half_t)a2[i];
      a[12 + i] = (half_t)a3[i];
    }

    const half_t* bbase = BT + (size_t)m * N_NODES + k0 + hi * 16;
#pragma unroll
    for (int g = 0; g < NT; g += 4) {
      v16h b[4];
#pragma unroll
      for (int t = 0; t < 4; ++t)
        b[t] = *(const v16h*)(bbase + (size_t)(g + t) * 16 * N_NODES);
      // Fence: the 8 loads of this group issue before the first WMMA, so
      // waits are staggered across 4 matrix ops (1 latency exposure / group,
      // only ~32 extra live VGPRs -> no spills).
      __builtin_amdgcn_sched_barrier(0);
#pragma unroll
      for (int t = 0; t < 4; ++t)
        acc[g + t] = __builtin_amdgcn_wmma_f32_16x16x32_f16(
            false, a, false, b[t], (short)0, acc[g + t], false, false);
    }
  }

  // ---- deterministic cross-wave tree reduction through LDS: 8 -> 4 -> 2 -> 1
#pragma unroll
  for (int s = 4; s >= 1; s >>= 1) {
    if (wave >= s && wave < 2 * s) {
      float* slot = red + ((size_t)((wave - s) * 32 + lane)) * (NT * 8);
#pragma unroll
      for (int t = 0; t < NT; ++t)
        *(v8f*)(slot + t * 8) = acc[t];
    }
    __syncthreads();
    if (wave < s) {
      const float* slot = red + ((size_t)(wave * 32 + lane)) * (NT * 8);
#pragma unroll
      for (int t = 0; t < NT; ++t)
        acc[t] += *(const v8f*)(slot + t * 8);
    }
    __syncthreads();
  }

  // ---- wave 0: ReLU + store
  if (wave == 0) {
#pragma unroll
    for (int t = 0; t < NT; ++t) {
#pragma unroll
      for (int r = 0; r < 8; ++r) {
        float v = acc[t][r];
        v = v > 0.f ? v : 0.f;
        const size_t row = (size_t)(row0 + r + hi * 8);
        const int    col = t * 16 + m;
        if (STORE_HALF)
          ((half_t*)outp)[row * DOUT + col] = (half_t)v;
        else
          ((float*)outp)[row * DOUT + col] = v;
      }
    }
  }
}

// ---------------------------------------------------------------------------
extern "C" void kernel_launch(void* const* d_in, const int* in_sizes, int n_in,
                              void* d_out, int out_size, void* d_ws, size_t ws_size,
                              hipStream_t stream) {
  const float* adj = (const float*)d_in[0];   // [8192, 8192] fp32
  const float* X   = (const float*)d_in[1];   // [8192, 64]   fp32
  const float* W1  = (const float*)d_in[2];   // [64, 128]    fp32
  const float* W2  = (const float*)d_in[3];   // [128, 256]   fp32

  char* ws = (char*)d_ws;
  half_t* T1T = (half_t*)ws;                                      // 2 MiB [128][8192]
  half_t* H1  = (half_t*)(ws + (size_t)D_HID * N_NODES * 2);      // 2 MiB [8192][128]
  half_t* T2T = (half_t*)(ws + (size_t)D_HID * N_NODES * 4);      // 4 MiB [256][8192]

  // Layer 1: T1 = X@W1 (fp16,T) ; H1 = relu(adj@T1) (fp16)
  k_xw1<<<(N_NODES * D_HID) / 256, 256, 0, stream>>>(X, W1, T1T);
  k_prop<D_HID, true><<<N_NODES / 16, 256, 0, stream>>>(adj, T1T, (void*)H1);

  // Layer 2: T2 = H1@W2 (fp16,T) ; out = relu(adj@T2) (fp32)
  k_hw2<<<(N_NODES * D_OUTC) / 256, 256, 0, stream>>>(H1, W2, T2T);
  k_prop<D_OUTC, false><<<N_NODES / 16, 256, 0, stream>>>(adj, T2T, d_out);
}
